// AttentionLayer_77867757077081
// MI455X (gfx1250) — matrix operand order repaired and checked by reading, not compiled, not run on hardware
//
#include <hip/hip_runtime.h>

typedef __attribute__((ext_vector_type(16))) _Float16 v16h;
typedef __attribute__((ext_vector_type(8)))  float    v8f;

namespace {
constexpr int kF = 40;
constexpr int kE = 64;
constexpr int kU = 64;
constexpr int kP = kF * (kF - 1) / 2;      // 780 pairs
constexpr int kPpad = 784;                 // 49 * 16
constexpr int kMT = 49;                    // 16-pair M tiles
constexpr int kThreads = 256;
constexpr int kWaves = kThreads / 32;      // wave32
constexpr int kBatchPerBlock = 8;
constexpr int kXStride = kE + 1;           // 65: LDS bank-conflict padding
}

union V16H { v16h v; _Float16 h[16]; };
union V8F  { v8f  v; float    f[8];  };

__global__ __launch_bounds__(kThreads)
void afm_attention_kernel(const float* __restrict__ x,      // [B,40,64]
                          const float* __restrict__ W,      // [64,64]
                          const float* __restrict__ bias,   // [64]
                          const float* __restrict__ projh,  // [64,1]
                          const float* __restrict__ projp,  // [64,1]
                          float* __restrict__ out,          // [B]
                          int B)
{
  __shared__ float xs[kF * kXStride];      // 10.4 KB, stride-65
  __shared__ float wlds[kE * kU];          // 16 KB staging for W
  __shared__ float hls[kU];
  __shared__ float bls[kU];
  __shared__ float pls[kE];
  __shared__ unsigned char ri8[kPpad];
  __shared__ unsigned char ci8[kPpad];
  __shared__ float scores[kP];
  __shared__ float dvals[kP];
  __shared__ float redA[kWaves];
  __shared__ float redB[kWaves];

  const int tid  = threadIdx.x;
  const int lane = tid & 31;
  const int wave = tid >> 5;
  const int nlo  = lane & 15;
  const bool hiHalf = lane >= 16;

  // ---------------- one-time per-block setup ----------------
  for (int i = tid; i < kE * kU; i += kThreads) wlds[i] = W[i];
  for (int i = tid; i < kU; i += kThreads) { hls[i] = projh[i]; bls[i] = bias[i]; }
  for (int i = tid; i < kE; i += kThreads) pls[i] = projp[i];
  for (int p = tid; p < kPpad; p += kThreads) {
    int r = 0, rem = p;
    while (r < kF - 2 && rem >= (kF - 1 - r)) { rem -= (kF - 1 - r); ++r; }
    int c = r + 1 + rem;
    if (p >= kP) { r = 0; c = 1; }         // pad rows: masked at score write
    ri8[p] = (unsigned char)r; ci8[p] = (unsigned char)c;
  }
  __syncthreads();

  // ---- build f16 B-fragments in registers (batch-invariant) ----
  // B 32x16 in this code's contiguous k order: lanes 0-15 hold K 0..15 (2 packed per VGPR),
  // lanes 16-31 hold K 16..31.  kk selects K block 0/32.
  V16H bf[2][4];
  {
    const int kofs = hiHalf ? 16 : 0;
    #pragma unroll
    for (int kk = 0; kk < 2; ++kk) {
      #pragma unroll
      for (int nt = 0; nt < 4; ++nt) {
        const int u = nt * 16 + nlo;
        #pragma unroll
        for (int v = 0; v < 8; ++v) {
          const int k0 = kk * 32 + kofs + 2 * v;
          bf[kk][nt].h[2 * v]     = (_Float16)wlds[k0 * kU + u];
          bf[kk][nt].h[2 * v + 1] = (_Float16)wlds[(k0 + 1) * kU + u];
        }
      }
    }
  }

  const long long b0 = (long long)blockIdx.x * kBatchPerBlock;
  for (int bi = 0; bi < kBatchPerBlock; ++bi) {
    const long long b = b0 + bi;
    if (b >= B) break;                     // uniform across block
    __syncthreads();                       // protect xs/scores/dvals reuse

    const float* xb = x + b * (long long)(kF * kE);
    for (int i = tid; i < kF * kE; i += kThreads)
      xs[(i >> 6) * kXStride + (i & 63)] = xb[i];
    if (bi + 1 < kBatchPerBlock && (b + 1) < (long long)B) {
      const float* xn = xb + kF * kE;      // prefetch next batch into caches
      for (int i = tid * 32; i < kF * kE; i += kThreads * 32)
        __builtin_prefetch(xn + i, 0, 1);
    }
    __syncthreads();

    // ---------------- scores via WMMA ----------------
    for (int mt = wave; mt < kMT; mt += kWaves) {
      const int p0 = mt * 16;
      const int prow = p0 + nlo;
      const float* xr = &xs[ri8[prow] * kXStride];
      const float* xc = &xs[ci8[prow] * kXStride];
      // A 16x32 (matching B): lanes 0-15 K 0..15, lanes 16-31 K 16..31
      const int kbase = hiHalf ? 16 : 0;
      V16H a[2];
      #pragma unroll
      for (int kk = 0; kk < 2; ++kk) {
        #pragma unroll
        for (int v = 0; v < 8; ++v) {
          const int k = kk * 32 + kbase + 2 * v;
          a[kk].h[2 * v]     = (_Float16)(xr[k]     * xc[k]);
          a[kk].h[2 * v + 1] = (_Float16)(xr[k + 1] * xc[k + 1]);
        }
      }
      float sacc[8] = {0.f, 0.f, 0.f, 0.f, 0.f, 0.f, 0.f, 0.f};
      #pragma unroll
      for (int nt = 0; nt < 4; ++nt) {
        v8f c = {};
        c = __builtin_amdgcn_wmma_f32_16x16x32_f16(false, a[0].v, false, bf[0][nt].v,
                                                   (short)0, c, false, false);
        c = __builtin_amdgcn_wmma_f32_16x16x32_f16(false, a[1].v, false, bf[1][nt].v,
                                                   (short)0, c, false, false);
        // fuse relu(.+bias)*h and accumulate per row
        const int u = nt * 16 + nlo;
        const float hu = hls[u], bu = bls[u];
        V8F cv; cv.v = c;
        #pragma unroll
        for (int r = 0; r < 8; ++r) {
          float t = cv.f[r] + bu;
          t = t > 0.f ? t : 0.f;
          sacc[r] += t * hu;
        }
      }
      // reduce over u within each 16-lane half (xor masks stay in-half)
      #pragma unroll
      for (int r = 0; r < 8; ++r) {
        float s = sacc[r];
        s += __shfl_xor(s, 1, 32);
        s += __shfl_xor(s, 2, 32);
        s += __shfl_xor(s, 4, 32);
        s += __shfl_xor(s, 8, 32);
        sacc[r] = s;
      }
      if (nlo == 0) {
        const int rb = p0 + (hiHalf ? 8 : 0);
        #pragma unroll
        for (int r = 0; r < 8; ++r) {
          const int p = rb + r;
          if (p < kP) scores[p] = sacc[r];
        }
      }
    }

    // ---------------- d_p = ip[p,:] . projection_p (f32) ----------------
    for (int p = tid; p < kP; p += kThreads) {
      const float* xr = &xs[ri8[p] * kXStride];
      const float* xc = &xs[ci8[p] * kXStride];
      float d = 0.f;
      #pragma unroll 8
      for (int e = 0; e < kE; ++e) d = fmaf(xr[e] * xc[e], pls[e], d);
      dvals[p] = d;
    }
    __syncthreads();   // scores + dvals complete for all waves

    // ---------------- softmax-weighted reduction ----------------
    float m = -3.0e38f;
    for (int p = tid; p < kP; p += kThreads) m = fmaxf(m, scores[p]);
    #pragma unroll
    for (int o = 16; o > 0; o >>= 1) m = fmaxf(m, __shfl_xor(m, o, 32));
    if (lane == 0) redA[wave] = m;
    __syncthreads();
    float mall = redA[0];
    #pragma unroll
    for (int w = 1; w < kWaves; ++w) mall = fmaxf(mall, redA[w]);
    __syncthreads();   // everyone done reading redA before reuse

    float se = 0.f, sd = 0.f;
    for (int p = tid; p < kP; p += kThreads) {
      const float ev = __expf(scores[p] - mall);
      se += ev;
      sd = fmaf(ev, dvals[p], sd);
    }
    #pragma unroll
    for (int o = 16; o > 0; o >>= 1) {
      se += __shfl_xor(se, o, 32);
      sd += __shfl_xor(sd, o, 32);
    }
    if (lane == 0) { redA[wave] = se; redB[wave] = sd; }
    __syncthreads();
    if (tid == 0) {
      float SE = 0.f, SD = 0.f;
      #pragma unroll
      for (int w = 0; w < kWaves; ++w) { SE += redA[w]; SD += redB[w]; }
      out[b] = SD / SE;
    }
  }
}

extern "C" void kernel_launch(void* const* d_in, const int* in_sizes, int n_in,
                              void* d_out, int out_size, void* d_ws, size_t ws_size,
                              hipStream_t stream) {
  const float* x  = (const float*)d_in[0];   // [B,40,64]
  const float* W  = (const float*)d_in[1];   // [64,64]
  const float* bb = (const float*)d_in[2];   // [64]
  const float* ph = (const float*)d_in[3];   // [64,1]
  const float* pp = (const float*)d_in[4];   // [64,1]
  float* out = (float*)d_out;                // [B,1]

  const int B = in_sizes[0] / (kF * kE);
  const int blocks = (B + kBatchPerBlock - 1) / kBatchPerBlock;
  afm_attention_kernel<<<blocks, kThreads, 0, stream>>>(x, W, bb, ph, pp, out, B);
}
